// Beam_59605556134005
// MI455X (gfx1250) — matrix operand order repaired and checked by reading, not compiled, not run on hardware
//
#include <hip/hip_runtime.h>
#include <hip/hip_bf16.h>

// ---------------------------------------------------------------------------
// Beam search for MI455X (gfx1250).
// Heavy op: per-step GEMM  [128 x 512] x [512 x 32000]  -> bf16 WMMA
// (v_wmma_f32_16x16x32_bf16) with f32 accumulate; W pre-packed to bf16
// fragment layout once (32MB, L2-resident). B fragments staged through LDS
// with gfx1250 GLOBAL_LOAD_ASYNC_TO_LDS_B128 (ASYNCcnt-tracked) and shared
// by the 8 M-tile waves of each block; 4 N-tiles per block to amortize
// A-fragment traffic.
// ---------------------------------------------------------------------------

#define BATCH   32
#define SEQLEN  32
#define KBEAM   4
#define VOCAB   32000
#define DDIM    512
#define NROWS   (BATCH * KBEAM)   // 128
#define BIG_NEG (-1.0e9f)

typedef __bf16 v16bf __attribute__((ext_vector_type(16)));
typedef float  v8f   __attribute__((ext_vector_type(8)));
typedef int    v4i   __attribute__((ext_vector_type(4)));

typedef __attribute__((address_space(1))) v4i* gptr_v4i;
typedef __attribute__((address_space(3))) v4i* lptr_v4i;

#if defined(__has_builtin)
#if __has_builtin(__builtin_amdgcn_global_load_async_to_lds_b128)
#define HAVE_ASYNC_LDS 1
#endif
#endif

// ---------------------------------------------------------------------------
// Pack W [DDIM x VOCAB] f32 -> bf16 WMMA B-fragments.
// Fragment (nt, kc): 32 lanes x 16 bf16, lane l covers column n = nt*16+(l&15),
// K range kc*32 + (l>=16 ? 16 : 0) + e  (e = 0..15).
// Fragments stored contiguously, index = nt*16 + kc, each 512 bf16.
// ---------------------------------------------------------------------------
__global__ void pack_W_kernel(const float* __restrict__ W, __bf16* __restrict__ Wp) {
    int frag = blockIdx.x * 8 + (threadIdx.x >> 5);   // 0 .. 31999
    int lane = threadIdx.x & 31;
    int nt = frag >> 4, kc = frag & 15;
    int n = nt * 16 + (lane & 15);
    int kbase = kc * 32 + ((lane >= 16) ? 16 : 0);
    __bf16* dst = Wp + (size_t)frag * 512 + lane * 16;
#pragma unroll
    for (int e = 0; e < 16; ++e)
        dst[e] = (__bf16)W[(size_t)(kbase + e) * VOCAB + n];
}

// ---------------------------------------------------------------------------
// Init beam state from x.
// ---------------------------------------------------------------------------
__global__ void init_state_kernel(const int* __restrict__ x, int* __restrict__ tgt,
                                  int* __restrict__ out, float* __restrict__ logp,
                                  float* __restrict__ score, int* __restrict__ flag) {
    int b = blockIdx.x;          // 32
    int t = threadIdx.x;         // 128 = KBEAM*SEQLEN
    int k = t >> 5, pos = t & 31;
    int v = x[b * SEQLEN + pos];
    tgt[(b * KBEAM + k) * SEQLEN + pos] = v;
    out[(b * KBEAM + k) * SEQLEN + pos] = v;
    if (pos == 0) {
        logp[b * KBEAM + k]  = (k == 0) ? 0.0f : BIG_NEG;
        score[b * KBEAM + k] = BIG_NEG;
        flag[b * KBEAM + k]  = 0;
    }
}

// ---------------------------------------------------------------------------
// Pack activations: emb[bk, :] = E[prev[bk], :] + ctx[b, :]  -> bf16 A-fragments.
// A fragment layout (contiguous k order, matches B): lane l, row M = l&15,
// K = kc*32 + (l>=16 ? 16 : 0) + e.
// Fragment index = mt*16 + kc (mt = 0..7, kc = 0..15), each 512 bf16.
// ---------------------------------------------------------------------------
__global__ void pack_A_kernel(const float* __restrict__ E, const float* __restrict__ ctx,
                              const int* __restrict__ tgt, __bf16* __restrict__ Ap,
                              int step) {
    int frag = blockIdx.x * 8 + (threadIdx.x >> 5);   // 0 .. 127
    int lane = threadIdx.x & 31;
    int mt = frag >> 4, kc = frag & 15;
    int bk = mt * 16 + (lane & 15);                   // 0 .. 127
    int b = bk >> 2;                                  // batch (KBEAM = 4)
    int prev = tgt[bk * SEQLEN + (step - 1)];
    int hi = (lane >= 16) ? 16 : 0;
    __bf16* dst = Ap + frag * 512 + lane * 16;
#pragma unroll
    for (int e = 0; e < 16; ++e) {
        int k = kc * 32 + hi + e;
        dst[e] = (__bf16)(E[(size_t)prev * DDIM + k] + ctx[b * DDIM + k]);
    }
}

// ---------------------------------------------------------------------------
// GEMM: z[128 x 32000] = A[128 x 512] * W[512 x 32000], bf16 WMMA, f32 acc.
// Block = 256 threads (8 waves). Wave w owns M-tile w. Each block covers
// 4 N-tiles (64 cols). B fragments (4 ntiles x 16 kc x 512 bf16 = 64KB)
// staged in LDS once (async global->LDS), shared by all 8 waves.
// ---------------------------------------------------------------------------
__global__ void __launch_bounds__(256) gemm_logits_kernel(
        const __bf16* __restrict__ Wp, const __bf16* __restrict__ Ap,
        float* __restrict__ z) {
    __shared__ __align__(16) __bf16 Bs[64 * 512];   // 64KB
    int nt0 = blockIdx.x * 4;

    // stage 64 contiguous fragments: Wp[nt0*16*512 ..)
    {
        const uint4* src = (const uint4*)(Wp + (size_t)nt0 * 16 * 512);
        uint4* dst = (uint4*)Bs;
#if defined(HAVE_ASYNC_LDS)
#pragma unroll
        for (int c = threadIdx.x; c < 4096; c += 256)   // 64KB / 16B
            __builtin_amdgcn_global_load_async_to_lds_b128(
                (gptr_v4i)(src + c), (lptr_v4i)(dst + c), 0, 0);
#if defined(__has_builtin) && __has_builtin(__builtin_amdgcn_s_wait_asynccnt)
        __builtin_amdgcn_s_wait_asynccnt(0);
#else
        asm volatile("s_wait_asynccnt 0" ::: "memory");
#endif
#else
#pragma unroll
        for (int c = threadIdx.x; c < 4096; c += 256)
            dst[c] = src[c];
#endif
    }
    __syncthreads();

    int wave = threadIdx.x >> 5;   // M-tile 0..7
    int lane = threadIdx.x & 31;
    v8f acc0 = {};
    v8f acc1 = {};
    v8f acc2 = {};
    v8f acc3 = {};
    const __bf16* ap = Ap + wave * 16 * 512 + lane * 16;
    const __bf16* bp = Bs + lane * 16;

#pragma unroll
    for (int kc = 0; kc < 16; ++kc) {
        v16bf a  = *(const v16bf*)(ap + kc * 512);
        v16bf b0 = *(const v16bf*)(bp + (0 * 16 + kc) * 512);
        v16bf b1 = *(const v16bf*)(bp + (1 * 16 + kc) * 512);
        v16bf b2 = *(const v16bf*)(bp + (2 * 16 + kc) * 512);
        v16bf b3 = *(const v16bf*)(bp + (3 * 16 + kc) * 512);
        acc0 = __builtin_amdgcn_wmma_f32_16x16x32_bf16(false, a, false, b0,
                                                       (short)0, acc0, false, false);
        acc1 = __builtin_amdgcn_wmma_f32_16x16x32_bf16(false, a, false, b1,
                                                       (short)0, acc1, false, false);
        acc2 = __builtin_amdgcn_wmma_f32_16x16x32_bf16(false, a, false, b2,
                                                       (short)0, acc2, false, false);
        acc3 = __builtin_amdgcn_wmma_f32_16x16x32_bf16(false, a, false, b3,
                                                       (short)0, acc3, false, false);
    }

    // C/D layout: VGPR e -> row M = e + (lane>=16 ? 8 : 0), col N = lane&15
    int mbase = wave * 16 + ((lane >= 16) ? 8 : 0);
    int ncol = lane & 15;
#pragma unroll
    for (int e = 0; e < 8; ++e) {
        size_t row = (size_t)(mbase + e) * VOCAB;
        z[row + (nt0 + 0) * 16 + ncol] = acc0[e];
        z[row + (nt0 + 1) * 16 + ncol] = acc1[e];
        z[row + (nt0 + 2) * 16 + ncol] = acc2[e];
        z[row + (nt0 + 3) * 16 + ncol] = acc3[e];
    }
}

// ---------------------------------------------------------------------------
// Per-row log-sum-exp over VOCAB. 128 blocks x 256 threads.
// ---------------------------------------------------------------------------
__global__ void row_lse_kernel(const float* __restrict__ z, float* __restrict__ lse) {
    __shared__ float red[256];
    int r = blockIdx.x;
    const float* zr = z + (size_t)r * VOCAB;
    float m = -INFINITY;
    for (int v = threadIdx.x; v < VOCAB; v += 256) m = fmaxf(m, zr[v]);
    red[threadIdx.x] = m;
    __syncthreads();
    for (int s = 128; s > 0; s >>= 1) {
        if (threadIdx.x < s) red[threadIdx.x] = fmaxf(red[threadIdx.x], red[threadIdx.x + s]);
        __syncthreads();
    }
    m = red[0];
    __syncthreads();
    float sum = 0.0f;
    for (int v = threadIdx.x; v < VOCAB; v += 256) sum += __expf(zr[v] - m);
    red[threadIdx.x] = sum;
    __syncthreads();
    for (int s = 128; s > 0; s >>= 1) {
        if (threadIdx.x < s) red[threadIdx.x] += red[threadIdx.x + s];
        __syncthreads();
    }
    if (threadIdx.x == 0) lse[r] = m + __logf(red[0]);
}

// ---------------------------------------------------------------------------
// Top-8 of y = logp[k] + z[k,v] - lse[k] over flattened (k, v) per batch.
// 32 blocks x 256 threads; per-thread sorted top-8 + LDS argmax rounds.
// ---------------------------------------------------------------------------
__global__ void topk8_kernel(const float* __restrict__ z, const float* __restrict__ logp,
                             const float* __restrict__ lse, float* __restrict__ topv,
                             int* __restrict__ topi) {
    __shared__ float sval[2048];
    __shared__ int   sidx[2048];
    __shared__ float rv[256];
    __shared__ int   ri[256];
    __shared__ int   rp[256];
    int b = blockIdx.x, tid = threadIdx.x;

    float bv[8]; int bi[8];
#pragma unroll
    for (int j = 0; j < 8; ++j) { bv[j] = -INFINITY; bi[j] = 0x7fffffff; }

    for (int k = 0; k < KBEAM; ++k) {
        int r = b * KBEAM + k;
        float base = logp[r] - lse[r];
        const float* zr = z + (size_t)r * VOCAB;
        for (int v = tid; v < VOCAB; v += 256) {
            float y = base + zr[v];
            if (y > bv[7]) {
                int idx = k * VOCAB + v;
                int j = 7;
                while (j > 0 && bv[j - 1] < y) { bv[j] = bv[j - 1]; bi[j] = bi[j - 1]; --j; }
                bv[j] = y; bi[j] = idx;
            }
        }
    }
#pragma unroll
    for (int j = 0; j < 8; ++j) { sval[tid * 8 + j] = bv[j]; sidx[tid * 8 + j] = bi[j]; }
    __syncthreads();

    for (int round = 0; round < 8; ++round) {
        float best = -INFINITY; int bestIdx = 0x7fffffff; int bestPos = tid * 8;
#pragma unroll
        for (int j = 0; j < 8; ++j) {
            float v = sval[tid * 8 + j];
            int   id = sidx[tid * 8 + j];
            if (v > best || (v == best && id < bestIdx)) { best = v; bestIdx = id; bestPos = tid * 8 + j; }
        }
        rv[tid] = best; ri[tid] = bestIdx; rp[tid] = bestPos;
        __syncthreads();
        for (int s = 128; s > 0; s >>= 1) {
            if (tid < s) {
                float vo = rv[tid + s]; int io = ri[tid + s];
                if (vo > rv[tid] || (vo == rv[tid] && io < ri[tid])) {
                    rv[tid] = vo; ri[tid] = io; rp[tid] = rp[tid + s];
                }
            }
            __syncthreads();
        }
        if (tid == 0) {
            topv[b * 8 + round] = rv[0];
            topi[b * 8 + round] = ri[0];
            sval[rp[0]] = -INFINITY;
        }
        __syncthreads();
    }
}

// ---------------------------------------------------------------------------
// Beam state update (tiny scalar work): 1 block, 1 thread per batch.
// ---------------------------------------------------------------------------
__global__ void beam_update_kernel(const float* __restrict__ topv, const int* __restrict__ topi,
                                   int* __restrict__ tgt, float* __restrict__ logp,
                                   int* __restrict__ out, float* __restrict__ score,
                                   int* __restrict__ flag, const int* __restrict__ end_id_p,
                                   int step) {
    int b = threadIdx.x;
    if (b >= BATCH) return;
    int end_id = end_id_p[0];

    float lp2k[8], lp_alive[8], sc[8];
    int beam[8], tok[8], fin[8];
    float n = (float)(step + 1);
    float pen = __powf((5.0f + n) / 6.0f, 0.6f);
#pragma unroll
    for (int j = 0; j < 8; ++j) {
        lp2k[j] = topv[b * 8 + j];
        int idx = topi[b * 8 + j];
        beam[j] = idx / VOCAB;
        tok[j]  = idx % VOCAB;
        fin[j]  = (tok[j] == end_id);
        lp_alive[j] = lp2k[j] + (fin[j] ? BIG_NEG : 0.0f);
        sc[j]       = lp2k[j] / pen + (fin[j] ? 0.0f : BIG_NEG);
    }

    int oldt[KBEAM][SEQLEN];
    for (int k = 0; k < KBEAM; ++k)
        for (int p = 0; p < SEQLEN; ++p)
            oldt[k][p] = tgt[(b * KBEAM + k) * SEQLEN + p];

    // top-4 alive beams (stable selection => lowest index wins ties)
    int used[8] = {0, 0, 0, 0, 0, 0, 0, 0};
    for (int r = 0; r < KBEAM; ++r) {
        int best = -1;
        for (int j = 0; j < 8; ++j)
            if (!used[j] && (best < 0 || lp_alive[j] > lp_alive[best])) best = j;
        used[best] = 1;
        int src = beam[best];
        for (int p = 0; p < SEQLEN; ++p)
            tgt[(b * KBEAM + r) * SEQLEN + p] = oldt[src][p];
        tgt[(b * KBEAM + r) * SEQLEN + step] = tok[best];
        logp[b * KBEAM + r] = lp_alive[best];
    }

    // top-4 finished beams
    int used2[8] = {0, 0, 0, 0, 0, 0, 0, 0};
    for (int r = 0; r < KBEAM; ++r) {
        int best = -1;
        for (int j = 0; j < 8; ++j)
            if (!used2[j] && (best < 0 || sc[j] > sc[best])) best = j;
        used2[best] = 1;
        int src = beam[best];
        for (int p = 0; p < SEQLEN; ++p)
            out[(b * KBEAM + r) * SEQLEN + p] = oldt[src][p];
        out[(b * KBEAM + r) * SEQLEN + step] = tok[best];
        score[b * KBEAM + r] = sc[best];
        flag[b * KBEAM + r]  = fin[best];
    }
}

// ---------------------------------------------------------------------------
// Final select: out = any_fin ? out : tgt ; score = any_fin ? score : logp.
// d_out = [B*K*L tokens as float][B*K scores].
// ---------------------------------------------------------------------------
__global__ void finalize_kernel(const int* __restrict__ tgt, const int* __restrict__ out,
                                const float* __restrict__ logp, const float* __restrict__ score,
                                const int* __restrict__ flag, float* __restrict__ dst) {
    __shared__ int anyf[BATCH];
    int tid = threadIdx.x;
    if (tid < BATCH) {
        int a = 0;
        for (int k = 0; k < KBEAM; ++k) a |= flag[tid * KBEAM + k];
        anyf[tid] = a;
    }
    __syncthreads();
    for (int t = tid; t < BATCH * KBEAM * SEQLEN; t += 256) {
        int b = t >> 7;   // KBEAM*SEQLEN = 128 per batch
        dst[t] = (float)(anyf[b] ? out[t] : tgt[t]);
    }
    for (int u = tid; u < BATCH * KBEAM; u += 256) {
        int b = u >> 2;
        dst[BATCH * KBEAM * SEQLEN + u] = anyf[b] ? score[u] : logp[u];
    }
}

// ---------------------------------------------------------------------------
extern "C" void kernel_launch(void* const* d_in, const int* in_sizes, int n_in,
                              void* d_out, int out_size, void* d_ws, size_t ws_size,
                              hipStream_t stream) {
    (void)in_sizes; (void)n_in; (void)out_size; (void)ws_size;
    const float* ctx   = (const float*)d_in[0];
    const float* E     = (const float*)d_in[1];
    const float* W     = (const float*)d_in[2];
    const int*   x     = (const int*)d_in[3];
    const int*   endid = (const int*)d_in[4];
    // d_in[5] = beam_size (compile-time KBEAM here)

    char* ws = (char*)d_ws;
    size_t off = 0;
    auto alloc = [&](size_t bytes) -> void* {
        void* p = ws + off;
        off += (bytes + 255) & ~(size_t)255;
        return p;
    };

    __bf16* Wp    = (__bf16*)alloc((size_t)VOCAB * DDIM * sizeof(__bf16));  // 32.75 MB
    __bf16* Ap    = (__bf16*)alloc((size_t)NROWS * DDIM * sizeof(__bf16));
    float*  z     = (float*)alloc((size_t)NROWS * VOCAB * sizeof(float));   // 16.4 MB
    float*  lse   = (float*)alloc(NROWS * sizeof(float));
    int*    tgt   = (int*)alloc(NROWS * SEQLEN * sizeof(int));
    int*    outb  = (int*)alloc(NROWS * SEQLEN * sizeof(int));
    float*  logp  = (float*)alloc(NROWS * sizeof(float));
    float*  score = (float*)alloc(NROWS * sizeof(float));
    int*    flag  = (int*)alloc(NROWS * sizeof(int));
    float*  topv  = (float*)alloc(BATCH * 8 * sizeof(float));
    int*    topi  = (int*)alloc(BATCH * 8 * sizeof(int));

    pack_W_kernel<<<VOCAB * 16 / 8, 256, 0, stream>>>(W, Wp);
    init_state_kernel<<<BATCH, KBEAM * SEQLEN, 0, stream>>>(x, tgt, outb, logp, score, flag);

    for (int i = 1; i < SEQLEN; ++i) {
        pack_A_kernel<<<16, 256, 0, stream>>>(E, ctx, tgt, Ap, i);
        gemm_logits_kernel<<<VOCAB / 64, 256, 0, stream>>>(Wp, Ap, z);
        row_lse_kernel<<<NROWS, 256, 0, stream>>>(z, lse);
        topk8_kernel<<<BATCH, 256, 0, stream>>>(z, logp, lse, topv, topi);
        beam_update_kernel<<<1, 32, 0, stream>>>(topv, topi, tgt, logp, outb, score,
                                                 flag, endid, i);
    }
    finalize_kernel<<<1, 256, 0, stream>>>(tgt, outb, logp, score, flag, (float*)d_out);
}
